// EuclidDistance_Assign_Module_32066225832561
// MI455X (gfx1250) — matrix operand order repaired and checked by reading, not compiled, not run on hardware
//
#include <hip/hip_runtime.h>

// ---------------------------------------------------------------------------
// MI455X (gfx1250) fused LayerNorm + Euclid-cdist + NegSoftAssign + reconstruct
//   outputs (concatenated f32): x_distance (M,K), x_distance_assign (M,K),
//                               x_rec (M,D)
// Strategy: bandwidth-bound -> single pass over x; GEMMs in bf16 WMMA
// (v_wmma_f32_16x16x32_bf16), f32 accumulate. One workgroup = 16 rows x K=512.
// Cluster-table tiles are streamed into LDS with the CDNA5 async-to-LDS DMA
// path (global_load_async_to_lds_b128 + s_wait_asynccnt), double-buffered so
// the next chunk's DMA overlaps the current chunk's WMMAs.
// ---------------------------------------------------------------------------

typedef __attribute__((ext_vector_type(16))) __bf16 v16bf;
typedef __attribute__((ext_vector_type(8)))  float  v8f;

constexpr int   Mtot  = 131072;
constexpr int   Dd    = 256;    // feature dim
constexpr int   Kk    = 512;    // clusters
constexpr float EPSV  = 1e-5f;
constexpr float ALPHA = 32.0f;

// Async copy of one 16-byte block global -> LDS (per lane), ASYNCcnt-tracked.
__device__ __forceinline__ void async_ld16(unsigned lds_addr, const void* gaddr) {
  asm volatile("global_load_async_to_lds_b128 %0, %1, off"
               :: "v"(lds_addr), "v"(gaddr)
               : "memory");
}
#define WAIT_ASYNCCNT(n) asm volatile("s_wait_asynccnt %0" :: "i"(n) : "memory")

// ------------------------- cluster prep ------------------------------------
__global__ __launch_bounds__(256) void prep_clusters_kernel(
    const float* __restrict__ cc, __bf16* __restrict__ c_bf,
    float* __restrict__ c_sq) {
  const int k = blockIdx.x;   // 0..511
  const int t = threadIdx.x;  // 0..255 (== D)
  float v = cc[(size_t)k * Dd + t];
  c_bf[(size_t)k * Dd + t] = (__bf16)v;
  __shared__ float red[256];
  red[t] = v * v;
  __syncthreads();
  for (int s = 128; s > 0; s >>= 1) {
    if (t < s) red[t] += red[t + s];
    __syncthreads();
  }
  if (t == 0) c_sq[k] = red[0];
}

// ------------------------- fused main kernel -------------------------------
// LDS layout (overlaid regions, ~121 KB total):
//   [OFF_X ] 16 KB : sX   f32 [16][256]  (phase 1)  == sAsn bf16[16][512] (phase 3+)
//   [OFF_A ]  8 KB : sA   bf16[16][256]  (xn, phases 1-2)
//   [OFF_B ] 64 KB : 2x sB bf16[512][32] (phase 2)  == 2x sC2 bf16[32][256] (phase 4)
//   [OFF_D ] 32 KB : sD   f32 [16][512]  (phases 2-3)
//   [OFF_R ] ~1 KB : sRed f32 [16][17]
//   [OFF_S ] 256 B : sStat f32[16][4]
constexpr int OFF_X = 0;
constexpr int OFF_A = 16384;
constexpr int OFF_B = 24576;     // buf stride 32768 (phase 2) / 16384 (phase 4)
constexpr int OFF_D = 90112;
constexpr int OFF_R = 122880;
constexpr int OFF_S = 123968;
constexpr int SMEM_BYTES = 124224;

__global__ __launch_bounds__(256) void fused_lncdist_kernel(
    const float* __restrict__ x,
    const float* __restrict__ lnw,
    const float* __restrict__ lnb,
    const __bf16* __restrict__ c_bf,
    const float* __restrict__ c_sq,
    float* __restrict__ out_d,
    float* __restrict__ out_assign,
    float* __restrict__ out_rec) {
  __shared__ __align__(16) char smem[SMEM_BYTES];
  float  (*sX)[256]   = reinterpret_cast<float  (*)[256]>(smem + OFF_X);
  __bf16 (*sAsn)[512] = reinterpret_cast<__bf16 (*)[512]>(smem + OFF_X);
  __bf16 (*sA)[256]   = reinterpret_cast<__bf16 (*)[256]>(smem + OFF_A);
  float  (*sD)[512]   = reinterpret_cast<float  (*)[512]>(smem + OFF_D);
  float  (*sRed)[17]  = reinterpret_cast<float  (*)[17] >(smem + OFF_R);
  float  (*sStat)[4]  = reinterpret_cast<float  (*)[4]  >(smem + OFF_S);

  const int    t    = threadIdx.x;     // 0..255 (8 wave32s)
  const int    r    = t >> 4;          // row 0..15 (reduction mapping)
  const int    c16  = t & 15;          // 16 threads per row
  const long   row0 = (long)blockIdx.x * 16;

  // LDS byte address of the B region (generic-LDS addr[31:0] == LDS offset)
  const unsigned ldsB = (unsigned)(uintptr_t)(smem + OFF_B);

  // Async stage of one GEMM1 B chunk (512 clusters x 32 feats bf16 = 32 KB).
  // 2048 16B blocks; 8 per thread; lane-consecutive blocks are contiguous.
  auto stage_B = [&](int kk, int buf) {
    const unsigned base = ldsB + (unsigned)buf * 32768u;
#pragma unroll
    for (int i = 0; i < 8; ++i) {
      int blk = i * 256 + t;        // 0..2047
      int row = blk >> 2;           // cluster 0..511
      int sub = blk & 3;            // 16B block within 64B row-chunk
      const char* src = (const char*)c_bf + ((long)row * Dd + kk * 32) * 2 + sub * 16;
      async_ld16(base + (unsigned)blk * 16u, src);
    }
  };
  // Async stage of one GEMM2 C chunk (32 cluster rows x 256 feats = 16 KB).
  auto stage_C2 = [&](int kk, int buf) {
    const unsigned base = ldsB + (unsigned)buf * 16384u;
#pragma unroll
    for (int i = 0; i < 4; ++i) {
      int blk = i * 256 + t;        // 0..1023
      int rowb = blk >> 5;          // 0..31
      int sub = blk & 31;           // 16B block within 512B row
      const char* src = (const char*)c_bf + ((long)(kk * 32 + rowb) * Dd) * 2 + sub * 16;
      async_ld16(base + (unsigned)blk * 16u, src);
    }
  };

  // Prefetch first B chunk: DMA overlaps the whole LayerNorm phase.
  stage_B(0, 0);

  // ---------------- Phase 1: load x tile + LayerNorm -> bf16 A, ||xn||^2 ---
  {
    const float4* src = reinterpret_cast<const float4*>(x + (row0 + r) * Dd + c16 * 16);
    float4* dst = reinterpret_cast<float4*>(&sX[r][c16 * 16]);
#pragma unroll
    for (int i = 0; i < 4; ++i) dst[i] = src[i];
  }
  __syncthreads();
  {  // mean
    float s = 0.f;
#pragma unroll
    for (int j = 0; j < 16; ++j) s += sX[r][c16 * 16 + j];
    sRed[r][c16] = s;
  }
  __syncthreads();
  if (c16 == 0) {
    float s = 0.f;
    for (int j = 0; j < 16; ++j) s += sRed[r][j];
    sStat[r][0] = s * (1.0f / Dd);
  }
  __syncthreads();
  {  // variance
    const float mean = sStat[r][0];
    float s = 0.f;
#pragma unroll
    for (int j = 0; j < 16; ++j) {
      float dlt = sX[r][c16 * 16 + j] - mean;
      s += dlt * dlt;
    }
    sRed[r][c16] = s;
  }
  __syncthreads();
  if (c16 == 0) {
    float s = 0.f;
    for (int j = 0; j < 16; ++j) s += sRed[r][j];
    float stddev = sqrtf(s * (1.0f / Dd));
    sStat[r][1] = 1.0f / (stddev + EPSV);   // eps added to std, per reference
  }
  __syncthreads();
  {  // normalize, affine, bf16 convert, ||xn||^2 partial
    const float mean = sStat[r][0];
    const float inv  = sStat[r][1];
    float sq = 0.f;
#pragma unroll
    for (int j = 0; j < 16; ++j) {
      int col = c16 * 16 + j;
      float xn = (sX[r][col] - mean) * inv * lnw[col] + lnb[col];
      sA[r][col] = (__bf16)xn;
      sq += xn * xn;
    }
    sRed[r][c16] = sq;
  }
  __syncthreads();
  if (c16 == 0) {
    float s = 0.f;
    for (int j = 0; j < 16; ++j) s += sRed[r][j];
    sStat[r][2] = s;   // ||xn||^2 for this row
  }
  __syncthreads();

  // ---------------- Phase 2: GEMM1 xn(16x256) @ C^T(256x512) --------------
  const int wave = t >> 5;
  const int lane = t & 31;
  const int half = lane >> 4;   // 0/1: which lane-group
  const int ln   = lane & 15;

  const v8f vzero = {0.f, 0.f, 0.f, 0.f, 0.f, 0.f, 0.f, 0.f};
  v8f acc[4];
#pragma unroll
  for (int i = 0; i < 4; ++i) acc[i] = vzero;

  for (int kk = 0; kk < 8; ++kk) {      // D = 8 chunks of 32, double-buffered
    if (kk + 1 < 8) {
      stage_B(kk + 1, (kk + 1) & 1);    // DMA next chunk while computing
      WAIT_ASYNCCNT(8);                 // in-order: <=8 left => chunk kk landed
    } else {
      WAIT_ASYNCCNT(0);
    }
    __syncthreads();                    // all waves' DMA visible
    __bf16 (*sB)[32] =
        reinterpret_cast<__bf16 (*)[32]>(smem + OFF_B + (kk & 1) * 32768);

    // A fragment: 16x32 bf16, documented CDNA5 lane layout
    union { v16bf v; __bf16 h[16]; } af;
    const int kb = kk * 32 + 8 * half;
#pragma unroll
    for (int j = 0; j < 8; ++j) {
      int kc = kb + 2 * j + (j >= 4 ? 8 : 0);
      af.h[2 * j]     = sA[ln][kc];
      af.h[2 * j + 1] = sA[ln][kc + 1];
    }
#pragma unroll
    for (int tc = 0; tc < 4; ++tc) {
      const int n = wave * 64 + tc * 16 + ln;   // cluster column 0..511
      union { v16bf v; __bf16 h[16]; } bfm;
#pragma unroll
      for (int j = 0; j < 8; ++j) {
        int kc = 2 * j + 8 * half + (j >= 4 ? 8 : 0);   // B(32x16) lane layout
        bfm.h[2 * j]     = sB[n][kc];
        bfm.h[2 * j + 1] = sB[n][kc + 1];
      }
      acc[tc] = __builtin_amdgcn_wmma_f32_16x16x32_bf16(
          false, af.v, false, bfm.v, (short)0, acc[tc], false, false);
    }
    __syncthreads();   // buffer reusable only after every wave consumed it
  }

  // distances: d = sqrt(||xn||^2 + ||c||^2 - 2*dot)
#pragma unroll
  for (int tc = 0; tc < 4; ++tc) {
    const int n = wave * 64 + tc * 16 + ln;
    const float csq = c_sq[n];
    union { v8f v; float f[8]; } au;
    au.v = acc[tc];
#pragma unroll
    for (int rr = 0; rr < 8; ++rr) {
      const int rowl = rr + 8 * half;           // C/D matrix lane layout
      float d2 = sStat[rowl][2] + csq - 2.0f * au.f[rr];
      float dd = sqrtf(fmaxf(d2, 0.0f));
      sD[rowl][n] = dd;
      out_d[(row0 + rowl) * Kk + n] = dd;
    }
  }

  // Prefetch first GEMM2 chunk into buf0 (old phase-2 buf0 area, now dead);
  // DMA overlaps the entire softmax phase.
  stage_C2(0, 0);
  __syncthreads();

  // ---------------- Phase 3: NegSoftAssign softmax over K ------------------
  {
    float s = 0.f, mn = 3.4e38f;
#pragma unroll
    for (int j = 0; j < 32; ++j) {
      float v = sD[r][c16 * 32 + j];
      s += v;
      mn = fminf(mn, v);
    }
    sRed[r][c16] = s;
    __syncthreads();
    if (c16 == 0) {
      float ss = 0.f;
      for (int j = 0; j < 16; ++j) ss += sRed[r][j];
      sStat[r][0] = ss * (1.0f / Kk);           // d_mean
    }
    __syncthreads();
    sRed[r][c16] = mn;
    __syncthreads();
    if (c16 == 0) {
      float m = 3.4e38f;
      for (int j = 0; j < 16; ++j) m = fminf(m, sRed[r][j]);
      sStat[r][1] = m;                          // d_min (=> max logit)
    }
    __syncthreads();
    const float scl  = -ALPHA / sStat[r][0];
    const float dmin = sStat[r][1];
    float se = 0.f;
#pragma unroll
    for (int j = 0; j < 32; ++j) {
      int col = c16 * 32 + j;
      float e = __expf(scl * (sD[r][col] - dmin));   // stable softmax
      sD[r][col] = e;
      se += e;
    }
    sRed[r][c16] = se;
    __syncthreads();
    if (c16 == 0) {
      float ss = 0.f;
      for (int j = 0; j < 16; ++j) ss += sRed[r][j];
      sStat[r][3] = ss;
    }
    __syncthreads();
    const float invse = 1.0f / sStat[r][3];
#pragma unroll
    for (int j = 0; j < 32; ++j) {
      int col = c16 * 32 + j;
      float a = sD[r][col] * invse;
      out_assign[(row0 + r) * Kk + col] = a;
      sAsn[r][col] = (__bf16)a;                 // overlays dead sX region
    }
  }
  __syncthreads();

  // ---------------- Phase 4: GEMM2 assign(16x512) @ C(512x256) -------------
  v8f acc2[2];
#pragma unroll
  for (int i = 0; i < 2; ++i) acc2[i] = vzero;

  for (int kk = 0; kk < 16; ++kk) {     // K = 16 chunks of 32, double-buffered
    if (kk + 1 < 16) {
      stage_C2(kk + 1, (kk + 1) & 1);
      WAIT_ASYNCCNT(4);                 // in-order: <=4 left => chunk kk landed
    } else {
      WAIT_ASYNCCNT(0);
    }
    __syncthreads();
    __bf16 (*sC2)[256] =
        reinterpret_cast<__bf16 (*)[256]>(smem + OFF_B + (kk & 1) * 16384);

    union { v16bf v; __bf16 h[16]; } af;
    const int kb = kk * 32 + 8 * half;
#pragma unroll
    for (int j = 0; j < 8; ++j) {
      int kc = kb + 2 * j + (j >= 4 ? 8 : 0);
      af.h[2 * j]     = sAsn[ln][kc];
      af.h[2 * j + 1] = sAsn[ln][kc + 1];
    }
#pragma unroll
    for (int tc = 0; tc < 2; ++tc) {
      const int n = wave * 32 + tc * 16 + ln;   // feature column 0..255
      union { v16bf v; __bf16 h[16]; } bfm;
#pragma unroll
      for (int j = 0; j < 8; ++j) {
        int kc = 2 * j + 8 * half + (j >= 4 ? 8 : 0);
        bfm.h[2 * j]     = sC2[kc][n];
        bfm.h[2 * j + 1] = sC2[kc + 1][n];
      }
      acc2[tc] = __builtin_amdgcn_wmma_f32_16x16x32_bf16(
          false, af.v, false, bfm.v, (short)0, acc2[tc], false, false);
    }
    __syncthreads();
  }

#pragma unroll
  for (int tc = 0; tc < 2; ++tc) {
    const int n = wave * 32 + tc * 16 + ln;
    union { v8f v; float f[8]; } au;
    au.v = acc2[tc];
#pragma unroll
    for (int rr = 0; rr < 8; ++rr) {
      const int rowl = rr + 8 * half;
      out_rec[(row0 + rowl) * Dd + n] = au.f[rr];
    }
  }
}

// --------------------------------------------------------------------------
extern "C" void kernel_launch(void* const* d_in, const int* in_sizes, int n_in,
                              void* d_out, int out_size, void* d_ws, size_t ws_size,
                              hipStream_t stream) {
  const float* x   = (const float*)d_in[0];
  const float* lnw = (const float*)d_in[1];
  const float* lnb = (const float*)d_in[2];
  const float* cc  = (const float*)d_in[3];
  float* out = (float*)d_out;

  // workspace: bf16 cluster table (256 KB) + per-cluster sq-norms (2 KB)
  __bf16* c_bf = (__bf16*)d_ws;
  float*  c_sq = (float*)((char*)d_ws + (size_t)Kk * Dd * sizeof(__bf16));

  prep_clusters_kernel<<<Kk, 256, 0, stream>>>(cc, c_bf, c_sq);

  float* out_d      = out;
  float* out_assign = out + (size_t)Mtot * Kk;
  float* out_rec    = out + 2ull * (size_t)Mtot * Kk;

  fused_lncdist_kernel<<<Mtot / 16, 256, 0, stream>>>(
      x, lnw, lnb, c_bf, c_sq, out_d, out_assign, out_rec);
}
